// TransformerLayer_11252814315999
// MI455X (gfx1250) — matrix operand order repaired and checked by reading, not compiled, not run on hardware
//
#include <hip/hip_runtime.h>
#include <hip/hip_bf16.h>
#include <math.h>

// ---------------------------------------------------------------------------
// Transformer layer for MI455X (gfx1250): bf16 WMMA everywhere, fp32 accum.
// ---------------------------------------------------------------------------

typedef __attribute__((ext_vector_type(8)))  float   v8f;
typedef __attribute__((ext_vector_type(16))) __bf16  v16bf;
typedef __attribute__((ext_vector_type(8)))  __bf16  v8bf;

#define D_MODEL  768
#define D_QKV    2304
#define D_FF     3072
#define N_HEADS  8
#define HEAD_DIM 96
#define SEQ      4096
#define BATCH    2
#define ROWS     (BATCH * SEQ)   // 8192

// ---- bf16 helpers (round-to-nearest-even) ---------------------------------
static __device__ __forceinline__ __bf16 f2bf(float f) {
  union { float f; unsigned u; } in; in.f = f;
  unsigned u = in.u;
  u += 0x7FFFu + ((u >> 16) & 1u);
  union { unsigned short s; __bf16 b; } out;
  out.s = (unsigned short)(u >> 16);
  return out.b;
}

static __device__ __forceinline__ v8f wmma_bf16(v16bf a, v16bf b, v8f c) {
  // 8 args: (neg_a, A, neg_b, B, c_mod, C, reuse_a, reuse_b)
  return __builtin_amdgcn_wmma_f32_16x16x32_bf16(false, a, false, b, (short)0, c,
                                                 false, false);
}

#define SHUF16(lo, hi) __builtin_shufflevector(lo, hi, 0,1,2,3,4,5,6,7,8,9,10,11,12,13,14,15)

// ---- DPP 16-lane butterfly reductions (no LDS, no waits) -------------------
// ctrl: 0xB1 = quad_perm(1,0,3,2)  -> lane ^= 1
//       0x4E = quad_perm(2,3,0,1)  -> lane ^= 2
//       0x141 = row_half_mirror    -> lane ^= 7  (covers xor-4 level)
//       0x140 = row_mirror         -> lane ^= 15 (covers xor-8 level)
template <int CTRL>
static __device__ __forceinline__ float dppf(float x) {
  return __builtin_bit_cast(
      float, __builtin_amdgcn_update_dpp(0, __builtin_bit_cast(int, x), CTRL,
                                         0xF, 0xF, true));
}
static __device__ __forceinline__ float red_max16(float x) {
  x = fmaxf(x, dppf<0xB1>(x));
  x = fmaxf(x, dppf<0x4E>(x));
  x = fmaxf(x, dppf<0x141>(x));
  x = fmaxf(x, dppf<0x140>(x));
  return x;
}
static __device__ __forceinline__ float red_sum16(float x) {
  x += dppf<0xB1>(x);
  x += dppf<0x4E>(x);
  x += dppf<0x141>(x);
  x += dppf<0x140>(x);
  return x;
}

// ---------------------------------------------------------------------------
// fp32 -> bf16 copy
// ---------------------------------------------------------------------------
__global__ void k_f32_to_bf16(const float* __restrict__ src,
                              __bf16* __restrict__ dst, int n) {
  int i = blockIdx.x * blockDim.x + threadIdx.x;
  if (i < n) dst[i] = f2bf(src[i]);
}

// ---------------------------------------------------------------------------
// Weight transpose: src [K][N] f32 (row-major)  ->  dst [N][K] bf16
// ---------------------------------------------------------------------------
__global__ void k_transpose_bf16(const float* __restrict__ src,
                                 __bf16* __restrict__ dst, int K, int N) {
  long i = (long)blockIdx.x * blockDim.x + threadIdx.x;
  long total = (long)K * N;
  if (i < total) {
    int n = (int)(i / K), k = (int)(i % K);
    dst[i] = f2bf(src[(long)k * N + n]);
  }
}

// ---------------------------------------------------------------------------
// Build V^T: vT[b][h][d][s] = qkv[b*SEQ+s][2*D_MODEL + h*HEAD_DIM + d]
// ---------------------------------------------------------------------------
__global__ void k_build_vT(const __bf16* __restrict__ qkv,
                           __bf16* __restrict__ vT) {
  const long total = (long)BATCH * N_HEADS * HEAD_DIM * SEQ;
  long i = (long)blockIdx.x * blockDim.x + threadIdx.x;
  if (i >= total) return;
  int s = (int)(i & (SEQ - 1));
  long t = i >> 12;                 // / SEQ
  int d = (int)(t % HEAD_DIM);
  t /= HEAD_DIM;
  int h = (int)(t & (N_HEADS - 1));
  int b = (int)(t >> 3);
  vT[i] = qkv[(long)(b * SEQ + s) * D_QKV + 2 * D_MODEL + h * HEAD_DIM + d];
}

// ---------------------------------------------------------------------------
// GEMM: D = A[M][K](bf16, row major) @ BT[N][K]^T (bf16) + bias
// One wave -> 16x64 strip (4 WMMA accumulators), register double-buffered:
// the k+32 fragments load while the k WMMAs execute. 4 waves/block along M.
// K compile-time. EPI: 0 = none, 1 = exact GELU.
// ---------------------------------------------------------------------------
template <int K, int EPI>
__global__ __launch_bounds__(128, 1)
void k_gemm_bf16(const __bf16* __restrict__ A,
                 const __bf16* __restrict__ BT,
                 const float* __restrict__ bias,
                 float* __restrict__ outF,
                 __bf16* __restrict__ outB,
                 int N) {
  const int lane = threadIdx.x & 31;
  const int wv   = threadIdx.x >> 5;
  const int m0   = (blockIdx.x * 4 + wv) * 16;
  const int n0   = blockIdx.y * 64;
  const int hi   = lane >> 4;
  const int lo   = lane & 15;

  const __bf16* arow = A + (size_t)(m0 + lo) * K + hi * 16;
  const __bf16* b0 = BT + (size_t)(n0 +  0 + lo) * K + hi * 16;
  const __bf16* b1 = BT + (size_t)(n0 + 16 + lo) * K + hi * 16;
  const __bf16* b2 = BT + (size_t)(n0 + 32 + lo) * K + hi * 16;
  const __bf16* b3 = BT + (size_t)(n0 + 48 + lo) * K + hi * 16;

  v8f acc0 = {}, acc1 = {}, acc2 = {}, acc3 = {};

  // prologue loads (k = 0)
  v8bf  alo = *(const v8bf*)(arow);
  v8bf  ahi = *(const v8bf*)(arow + 8);
  v16bf f0  = *(const v16bf*)(b0);
  v16bf f1  = *(const v16bf*)(b1);
  v16bf f2  = *(const v16bf*)(b2);
  v16bf f3  = *(const v16bf*)(b3);

#pragma unroll 4
  for (int k0 = 0; k0 < K; k0 += 32) {
    v16bf af = SHUF16(alo, ahi);
    v16bf c0 = f0, c1 = f1, c2 = f2, c3 = f3;
    // double-buffer: fetch next-k fragments (redundant reload on last iter)
    int kn = (k0 + 32 < K) ? (k0 + 32) : k0;
    if (kn + 256 < K) __builtin_prefetch(arow + kn + 256, 0, 1);
    alo = *(const v8bf*)(arow + kn);
    ahi = *(const v8bf*)(arow + kn + 8);
    f0  = *(const v16bf*)(b0 + kn);
    f1  = *(const v16bf*)(b1 + kn);
    f2  = *(const v16bf*)(b2 + kn);
    f3  = *(const v16bf*)(b3 + kn);
    acc0 = wmma_bf16(af, c0, acc0);
    acc1 = wmma_bf16(af, c1, acc1);
    acc2 = wmma_bf16(af, c2, acc2);
    acc3 = wmma_bf16(af, c3, acc3);
  }

  v8f accs[4] = {acc0, acc1, acc2, acc3};
#pragma unroll
  for (int j = 0; j < 4; ++j) {
    int cc = n0 + j * 16 + lo;
    float bv = bias ? bias[cc] : 0.0f;
#pragma unroll
    for (int r = 0; r < 8; ++r) {
      int rr = m0 + r + hi * 8;
      float v = accs[j][r] + bv;
      if (EPI == 1) v = 0.5f * v * (1.0f + erff(v * 0.70710678118654752f));
      if (outF) outF[(size_t)rr * N + cc] = v;
      if (outB) outB[(size_t)rr * N + cc] = f2bf(v);
    }
  }
}

// ---------------------------------------------------------------------------
// Flash attention: one wave handles one 16-row q-tile of one (b, h).
// Per 32-key step: scores (K frags staged 3+3), V loads issued early,
// DPP online softmax (no LDS traffic), P repack via 1KB LDS, 6 P.V WMMAs.
// launch_bounds(32,1): full VGPR budget, no scratch spills.
// ---------------------------------------------------------------------------
__global__ __launch_bounds__(32, 1)
void k_attention(const __bf16* __restrict__ qkv,
                 const __bf16* __restrict__ vT,
                 __bf16* __restrict__ attn_out) {
  __shared__ __bf16 pTile[16 * 32];

  const int lane = threadIdx.x & 31;
  const int hi = lane >> 4, lo = lane & 15;
  const int qt = blockIdx.x;   // 0..255
  const int h  = blockIdx.y;   // 0..7
  const int b  = blockIdx.z;   // 0..1
  const long rowBase = (long)b * SEQ;

  // Q fragments: 16 rows x 96 (3 K-steps of 32)
  const __bf16* qrow =
      qkv + (rowBase + qt * 16 + lo) * D_QKV + h * HEAD_DIM + hi * 16;
  v16bf qa0, qa1, qa2;
  {
    v8bf l0 = *(const v8bf*)(qrow +  0), h0 = *(const v8bf*)(qrow + 8);
    v8bf l1 = *(const v8bf*)(qrow + 32), h1 = *(const v8bf*)(qrow + 40);
    v8bf l2 = *(const v8bf*)(qrow + 64), h2 = *(const v8bf*)(qrow + 72);
    qa0 = SHUF16(l0, h0);
    qa1 = SHUF16(l1, h1);
    qa2 = SHUF16(l2, h2);
  }

  // running pointers (incremented per key block; no per-iter multiplies)
  const __bf16* kp0 =
      qkv + (rowBase + lo) * D_QKV + D_MODEL + h * HEAD_DIM + hi * 16;
  const __bf16* kp1 = kp0 + (size_t)16 * D_QKV;
  const __bf16* vp =
      vT + ((long)(b * N_HEADS + h) * HEAD_DIM + lo) * SEQ + hi * 16;

  float rmax[8], rsum[8];
  v8f o0 = {}, o1 = {}, o2 = {}, o3 = {}, o4 = {}, o5 = {};
#pragma unroll
  for (int r = 0; r < 8; ++r) { rmax[r] = -1e30f; rsum[r] = 0.0f; }
  const float scale = 0.10206207261596575f;  // 1/sqrt(96)

  for (int kt0 = 0; kt0 < SEQ; kt0 += 32) {
    // ---- score tile 0 (keys kt0..kt0+15): stage 3 K frags, 3 WMMAs ----
    v8f s0 = {}, s1 = {};
    {
      v16bf ka0 = *(const v16bf*)(kp0 +  0);
      v16bf ka1 = *(const v16bf*)(kp0 + 32);
      v16bf ka2 = *(const v16bf*)(kp0 + 64);
      s0 = wmma_bf16(qa0, ka0, s0);
      s0 = wmma_bf16(qa1, ka1, s0);
      s0 = wmma_bf16(qa2, ka2, s0);
    }
    // ---- score tile 1 (keys kt0+16..kt0+31) ----
    {
      v16bf kb0 = *(const v16bf*)(kp1 +  0);
      v16bf kb1 = *(const v16bf*)(kp1 + 32);
      v16bf kb2 = *(const v16bf*)(kp1 + 64);
      s1 = wmma_bf16(qa0, kb0, s1);
      s1 = wmma_bf16(qa1, kb1, s1);
      s1 = wmma_bf16(qa2, kb2, s1);
    }

    // ---- issue V-fragment loads now; latency hides under the softmax ----
    v16bf vf0 = *(const v16bf*)(vp + (size_t)0 * 16 * SEQ);
    v16bf vf1 = *(const v16bf*)(vp + (size_t)1 * 16 * SEQ);
    v16bf vf2 = *(const v16bf*)(vp + (size_t)2 * 16 * SEQ);
    v16bf vf3 = *(const v16bf*)(vp + (size_t)3 * 16 * SEQ);
    v16bf vf4 = *(const v16bf*)(vp + (size_t)4 * 16 * SEQ);
    v16bf vf5 = *(const v16bf*)(vp + (size_t)5 * 16 * SEQ);

    // ---- online softmax: DPP butterfly reductions across the 16 lanes ----
#pragma unroll
    for (int r = 0; r < 8; ++r) {
      float a = s0[r] * scale, c = s1[r] * scale;
      float m = red_max16(fmaxf(a, c));
      float nm = fmaxf(rmax[r], m);
      float corr = __expf(rmax[r] - nm);
      float p0 = __expf(a - nm), p1 = __expf(c - nm);
      float ps = red_sum16(p0 + p1);
      rsum[r] = rsum[r] * corr + ps;
      rmax[r] = nm;
      o0[r] *= corr; o1[r] *= corr; o2[r] *= corr;
      o3[r] *= corr; o4[r] *= corr; o5[r] *= corr;
      int rr = r + hi * 8;
      pTile[rr * 32 + lo]      = f2bf(p0);
      pTile[rr * 32 + 16 + lo] = f2bf(p1);
    }
    __syncthreads();

    // ---- repack P (C-layout -> A-layout) through LDS ----
    v16bf pf;
    {
      v8bf pl = *(const v8bf*)&pTile[lo * 32 + hi * 16];
      v8bf ph = *(const v8bf*)&pTile[lo * 32 + hi * 16 + 8];
      pf = SHUF16(pl, ph);
    }

    // ---- O += P @ V (d = 96 -> 6 column tiles) ----
    o0 = wmma_bf16(pf, vf0, o0);
    o1 = wmma_bf16(pf, vf1, o1);
    o2 = wmma_bf16(pf, vf2, o2);
    o3 = wmma_bf16(pf, vf3, o3);
    o4 = wmma_bf16(pf, vf4, o4);
    o5 = wmma_bf16(pf, vf5, o5);
    __syncthreads();

    kp0 += (size_t)32 * D_QKV;
    kp1 += (size_t)32 * D_QKV;
    vp  += 32;
  }

  // ---- normalize and store bf16 ----
  v8f oo[6] = {o0, o1, o2, o3, o4, o5};
#pragma unroll
  for (int j = 0; j < 6; ++j) {
#pragma unroll
    for (int r = 0; r < 8; ++r) {
      float v = oo[j][r] / rsum[r];
      int rr = qt * 16 + r + hi * 8;
      attn_out[(rowBase + rr) * D_MODEL + h * HEAD_DIM + j * 16 + lo] = f2bf(v);
    }
  }
}

// ---------------------------------------------------------------------------
// y = LayerNorm(a + bsrc) * g + beta ; writes fp32 (and optionally bf16)
// One 256-thread block per row of 768.
// ---------------------------------------------------------------------------
__global__ void k_add_layernorm(const float* __restrict__ a,
                                const float* __restrict__ bsrc,
                                const float* __restrict__ g,
                                const float* __restrict__ beta,
                                float* __restrict__ outF,
                                __bf16* __restrict__ outB) {
  __shared__ float red[256];
  const int row = blockIdx.x;
  const int tid = threadIdx.x;
  const float* ar = a + (size_t)row * D_MODEL;
  const float* br = bsrc + (size_t)row * D_MODEL;
  float v0 = ar[tid]       + br[tid];
  float v1 = ar[tid + 256] + br[tid + 256];
  float v2 = ar[tid + 512] + br[tid + 512];

  red[tid] = v0 + v1 + v2;
  __syncthreads();
  for (int st = 128; st > 0; st >>= 1) {
    if (tid < st) red[tid] += red[tid + st];
    __syncthreads();
  }
  float mu = red[0] * (1.0f / D_MODEL);
  __syncthreads();

  float d0 = v0 - mu, d1 = v1 - mu, d2 = v2 - mu;
  red[tid] = d0 * d0 + d1 * d1 + d2 * d2;
  __syncthreads();
  for (int st = 128; st > 0; st >>= 1) {
    if (tid < st) red[tid] += red[tid + st];
    __syncthreads();
  }
  float rstd = rsqrtf(red[0] * (1.0f / D_MODEL) + 1e-5f);

  float y0 = d0 * rstd * g[tid]       + beta[tid];
  float y1 = d1 * rstd * g[tid + 256] + beta[tid + 256];
  float y2 = d2 * rstd * g[tid + 512] + beta[tid + 512];
  float* orow = outF + (size_t)row * D_MODEL;
  orow[tid] = y0; orow[tid + 256] = y1; orow[tid + 512] = y2;
  if (outB) {
    __bf16* ob = outB + (size_t)row * D_MODEL;
    ob[tid] = f2bf(y0); ob[tid + 256] = f2bf(y1); ob[tid + 512] = f2bf(y2);
  }
}

// ---------------------------------------------------------------------------
// Host launcher
// ---------------------------------------------------------------------------
extern "C" void kernel_launch(void* const* d_in, const int* in_sizes, int n_in,
                              void* d_out, int out_size, void* d_ws,
                              size_t ws_size, hipStream_t stream) {
  (void)in_sizes; (void)n_in; (void)out_size; (void)ws_size;
  const float* x     = (const float*)d_in[0];
  const float* w_qkv = (const float*)d_in[1];
  const float* b_qkv = (const float*)d_in[2];
  const float* w_out = (const float*)d_in[3];
  const float* b_out = (const float*)d_in[4];
  const float* w_ff1 = (const float*)d_in[5];
  const float* b_ff1 = (const float*)d_in[6];
  const float* w_ff2 = (const float*)d_in[7];
  const float* b_ff2 = (const float*)d_in[8];
  const float* g1    = (const float*)d_in[9];
  const float* be1   = (const float*)d_in[10];
  const float* g2    = (const float*)d_in[11];
  const float* be2   = (const float*)d_in[12];
  float* out = (float*)d_out;

  size_t off = 0;
  auto carve = [&](size_t bytes) -> char* {
    char* p = (char*)d_ws + off;
    off += (bytes + 255) & ~(size_t)255;
    return p;
  };
  __bf16* xb    = (__bf16*)carve((size_t)ROWS * D_MODEL * 2);
  __bf16* wqkvT = (__bf16*)carve((size_t)D_QKV * D_MODEL * 2);
  __bf16* woutT = (__bf16*)carve((size_t)D_MODEL * D_MODEL * 2);
  __bf16* wff1T = (__bf16*)carve((size_t)D_FF * D_MODEL * 2);
  __bf16* wff2T = (__bf16*)carve((size_t)D_MODEL * D_FF * 2);
  __bf16* qkvb  = (__bf16*)carve((size_t)ROWS * D_QKV * 2);
  __bf16* vT    = (__bf16*)carve((size_t)BATCH * N_HEADS * HEAD_DIM * SEQ * 2);
  __bf16* attnb = (__bf16*)carve((size_t)ROWS * D_MODEL * 2);
  float*  projF = (float*)carve((size_t)ROWS * D_MODEL * 4);
  float*  y1F   = (float*)carve((size_t)ROWS * D_MODEL * 4);
  __bf16* y1b   = (__bf16*)carve((size_t)ROWS * D_MODEL * 2);
  __bf16* hb    = (__bf16*)carve((size_t)ROWS * D_FF * 2);

  // 1. downconvert activations & weights
  {
    int n = ROWS * D_MODEL;
    k_f32_to_bf16<<<(n + 255) / 256, 256, 0, stream>>>(x, xb, n);
  }
  k_transpose_bf16<<<((long)D_MODEL * D_QKV + 255) / 256, 256, 0, stream>>>(
      w_qkv, wqkvT, D_MODEL, D_QKV);
  k_transpose_bf16<<<((long)D_MODEL * D_MODEL + 255) / 256, 256, 0, stream>>>(
      w_out, woutT, D_MODEL, D_MODEL);
  k_transpose_bf16<<<((long)D_MODEL * D_FF + 255) / 256, 256, 0, stream>>>(
      w_ff1, wff1T, D_MODEL, D_FF);
  k_transpose_bf16<<<((long)D_FF * D_MODEL + 255) / 256, 256, 0, stream>>>(
      w_ff2, wff2T, D_FF, D_MODEL);

  dim3 blk(128);
  // 2. QKV projection: [8192,768] x [768,2304]
  k_gemm_bf16<D_MODEL, 0><<<dim3(ROWS / 64, D_QKV / 64), blk, 0, stream>>>(
      xb, wqkvT, b_qkv, nullptr, qkvb, D_QKV);

  // 3. V transpose for attention B-operands
  {
    long n = (long)BATCH * N_HEADS * HEAD_DIM * SEQ;
    k_build_vT<<<(n + 255) / 256, 256, 0, stream>>>(qkvb, vT);
  }

  // 4. flash attention
  k_attention<<<dim3(SEQ / 16, N_HEADS, BATCH), 32, 0, stream>>>(qkvb, vT, attnb);

  // 5. output projection
  k_gemm_bf16<D_MODEL, 0><<<dim3(ROWS / 64, D_MODEL / 64), blk, 0, stream>>>(
      attnb, woutT, b_out, projF, nullptr, D_MODEL);

  // 6. residual + LN1
  k_add_layernorm<<<ROWS, 256, 0, stream>>>(x, projF, g1, be1, y1F, y1b);

  // 7. FF1 + exact GELU
  k_gemm_bf16<D_MODEL, 1><<<dim3(ROWS / 64, D_FF / 64), blk, 0, stream>>>(
      y1b, wff1T, b_ff1, nullptr, hb, D_FF);

  // 8. FF2
  k_gemm_bf16<D_FF, 0><<<dim3(ROWS / 64, D_MODEL / 64), blk, 0, stream>>>(
      hb, wff2T, b_ff2, projF, nullptr, D_MODEL);

  // 9. residual + LN2 -> final fp32 output
  k_add_layernorm<<<ROWS, 256, 0, stream>>>(y1F, projF, g2, be2, out, nullptr);
}
